// SurfaceToVolumeCVAE_9268539424871
// MI455X (gfx1250) — matrix operand order repaired and checked by reading, not compiled, not run on hardware
//
#include <hip/hip_runtime.h>

// ---------------------------------------------------------------------------
// SurfaceToVolumeCVAE forward for gfx1250 (wave32, WMMA bf16 f32-accumulate)
// ---------------------------------------------------------------------------

#define DEVFN __device__ __forceinline__

constexpr int BB  = 4;      // batch
constexpr int NPT = 2048;   // points
constexpr int KK  = 20;     // knn
constexpr int ZD  = 512;    // z dim
constexpr int CD  = 1024;   // cond dim
constexpr int NF  = 4096;   // template points

typedef __attribute__((ext_vector_type(16))) __bf16 v16bf;
typedef __attribute__((ext_vector_type(8)))  float  v8f;

union Frag16 {
  v16bf v;
  unsigned int   u32[8];
  unsigned short u16[16];
};

DEVFN unsigned short f2bf(float f) {
  unsigned int u = __float_as_uint(f);
  u += 0x7FFFu + ((u >> 16) & 1u);           // round-to-nearest-even
  return (unsigned short)(u >> 16);
}
DEVFN float bf2f(unsigned short s) { return __uint_as_float(((unsigned int)s) << 16); }

// A fragment (16 rows x 32 k, bf16) from LDS row-major [16][pitch]
// Layout per ISA 7.12.2: lane<16 holds K {0..7,16..23}; lane>=16 holds K {8..15,24..31}
DEVFN v16bf load_a_frag(const unsigned short* sh, int pitch, int k0, int lane) {
  Frag16 f;
  const int m  = lane & 15;
  const int kh = (lane < 16) ? 0 : 8;
#pragma unroll
  for (int i = 0; i < 8; ++i) {
    int k = k0 + kh + ((i < 4) ? (2 * i) : (16 + 2 * (i - 4)));
    f.u16[2 * i]     = sh[m * pitch + k];
    f.u16[2 * i + 1] = sh[m * pitch + k + 1];
  }
  return f.v;
}

// B fragment (32 k x 16 cols, bf16) from pre-packed weights (8 dwords / lane)
DEVFN v16bf load_b_frag(const unsigned int* pk, int ct, int kt, int KT, int lane) {
  Frag16 f;
  const unsigned int* p = pk + ((size_t)(ct * KT + kt)) * 256 + lane * 8;
#pragma unroll
  for (int i = 0; i < 8; ++i) f.u32[i] = p[i];
  return f.v;
}

DEVFN v8f wmma_bf16(v16bf a, v16bf b, v8f c) {
  return __builtin_amdgcn_wmma_f32_16x16x32_bf16(false, a, false, b, (short)0, c, false, false);
}

// One group of NG=4 output-column tiles: each A fragment loaded once feeds 4 WMMAs.
// Bounded live set (4x8 acc + 8 A + 8 B VGPRs) -> no scratch spills.
DEVFN void gemm_group4(const unsigned short* sA, int pitch, const unsigned int* pk,
                       int ct0, int KT, int lane, v8f acc[4]) {
  for (int kt = 0; kt < KT; ++kt) {
    v16bf a = load_a_frag(sA, pitch, kt * 32, lane);
#pragma unroll
    for (int j = 0; j < 4; ++j)
      acc[j] = wmma_bf16(a, load_b_frag(pk, ct0 + j, kt, KT, lane), acc[j]);
  }
}

// ---------------------------------------------------------------------------
// Pack W (OutC x InC, fp32 row-major) into bf16 WMMA B-fragment tile order.
// ---------------------------------------------------------------------------
__global__ void pack_weights(const float* __restrict__ W, unsigned int* __restrict__ pk,
                             int OutC, int InC, int KT) {
  int t = blockIdx.x * blockDim.x + threadIdx.x;
  int total = (OutC / 16) * KT * 256;
  if (t >= total) return;
  int i    = t & 7;
  int lane = (t >> 3) & 31;
  int tile = t >> 8;
  int kt = tile % KT;
  int ct = tile / KT;
  int k = kt * 32 + ((lane >= 16) ? 8 : 0) + ((i < 4) ? (2 * i) : (16 + 2 * (i - 4)));
  int c = ct * 16 + (lane & 15);
  float v0 = (k     < InC) ? W[(size_t)c * InC + k]     : 0.f;
  float v1 = (k + 1 < InC) ? W[(size_t)c * InC + k + 1] : 0.f;
  pk[t] = (unsigned int)f2bf(v0) | ((unsigned int)f2bf(v1) << 16);
}

// ---------------------------------------------------------------------------
// kNN: one thread per point, row cached in LDS (padded stride: no bank conflicts)
// ---------------------------------------------------------------------------
template<int D>
__global__ void knn_kernel(const float* __restrict__ x, int* __restrict__ idx) {
  constexpr int DP = D + 1;                 // pad: stride % 64 banks != 0
  __shared__ float rows[32 * DP];
  const int lane = threadIdx.x;
  const int p = blockIdx.x * 32 + lane;      // b*NPT + n
  const int b = p / NPT;
  const int n = p % NPT;
  const float* xb = x + (size_t)b * NPT * D;
  float* rl = rows + lane * DP;
  for (int d = 0; d < D; ++d) rl[d] = xb[(size_t)n * D + d];

  float dk[KK]; int ik[KK];
#pragma unroll
  for (int i = 0; i < KK; ++i) { dk[i] = 3.4e38f; ik[i] = 0; }

  for (int m = 0; m < NPT; ++m) {
    float s = 0.f;
    for (int d = 0; d < D; ++d) {
      float df = rl[d] - xb[(size_t)m * D + d];
      s += df * df;
    }
    if (s < dk[KK - 1]) {
      dk[KK - 1] = s; ik[KK - 1] = m;
#pragma unroll
      for (int t = KK - 1; t > 0; --t) {
        if (dk[t] < dk[t - 1]) {
          float td = dk[t]; dk[t] = dk[t - 1]; dk[t - 1] = td;
          int   ti = ik[t]; ik[t] = ik[t - 1]; ik[t - 1] = ti;
        }
      }
    }
  }
#pragma unroll
  for (int i = 0; i < KK; ++i) idx[(size_t)p * KK + i] = ik[i];
}

// ---------------------------------------------------------------------------
// EdgeConv GEMM: rows = B*N*K, feat = [nbr-center | center] (2D ch), h = feat @ W^T
// ---------------------------------------------------------------------------
template<int D, int C>
__global__ void edgeconv_gemm(const float* __restrict__ x, const int* __restrict__ idx,
                              const unsigned int* __restrict__ pkW, float* __restrict__ h) {
  constexpr int IN2 = 2 * D;
  constexpr int INP = (IN2 + 31) & ~31;
  constexpr int KT  = INP / 32;
  constexpr int PIT = INP + 8;              // +4 dwords: bank-conflict-free rows
  __shared__ unsigned short sA[16 * PIT];
  const int lane = threadIdx.x;
  const int row0 = blockIdx.x * 16;

  for (int e = lane; e < 16 * INP; e += 32) {
    int m = e / INP, d = e % INP;
    int r  = row0 + m;
    int b  = r / (NPT * KK);
    int rn = r % (NPT * KK);
    int n  = rn / KK, kk = rn % KK;
    float v = 0.f;
    if (d < IN2) {
      const float* xb = x + (size_t)b * NPT * D;
      if (d < D) {
        int nb = idx[(size_t)(b * NPT + n) * KK + kk];
        v = xb[(size_t)nb * D + d] - xb[(size_t)n * D + d];
      } else {
        v = xb[(size_t)n * D + (d - D)];
      }
    }
    sA[m * PIT + d] = f2bf(v);
  }
  __syncthreads();

  for (int ct0 = 0; ct0 < C / 16; ct0 += 4) {
    if (ct0 + 4 < C / 16)
      __builtin_prefetch((const void*)(pkW + ((size_t)(ct0 + 4) * KT) * 256), 0, 1);
    v8f acc[4] = {};
    gemm_group4(sA, PIT, pkW, ct0, KT, lane, acc);
    int col = lane & 15, rb = (lane < 16) ? 0 : 8;
#pragma unroll
    for (int j = 0; j < 4; ++j)
#pragma unroll
      for (int r = 0; r < 8; ++r)
        h[(size_t)(row0 + rb + r) * C + (ct0 + j) * 16 + col] = acc[j][r];
  }
}

// Aggregation GEMM over concat(x1|x2|x3|x4) (960 ch) -> 512
__global__ void agg_gemm(const float* __restrict__ x1, const float* __restrict__ x2,
                         const float* __restrict__ x3, const float* __restrict__ x4,
                         const unsigned int* __restrict__ pkW, float* __restrict__ h) {
  constexpr int INC = 960;
  constexpr int KT  = 30;
  constexpr int PIT = INC + 8;
  __shared__ unsigned short sA[16 * PIT];
  const int lane = threadIdx.x;
  const int row0 = blockIdx.x * 16;   // row = b*NPT + n

  for (int e = lane; e < 16 * INC; e += 32) {
    int m = e / INC, d = e % INC;
    size_t bn = (size_t)(row0 + m);
    float v;
    if      (d < 64)  v = x1[bn * 64  + d];
    else if (d < 192) v = x2[bn * 128 + (d - 64)];
    else if (d < 448) v = x3[bn * 256 + (d - 192)];
    else              v = x4[bn * 512 + (d - 448)];
    sA[m * PIT + d] = f2bf(v);
  }
  __syncthreads();

  for (int ct0 = 0; ct0 < 32; ct0 += 4) {
    if (ct0 + 4 < 32)
      __builtin_prefetch((const void*)(pkW + ((size_t)(ct0 + 4) * KT) * 256), 0, 1);
    v8f acc[4] = {};
    gemm_group4(sA, PIT, pkW, ct0, KT, lane, acc);
    int col = lane & 15, rb = (lane < 16) ? 0 : 8;
#pragma unroll
    for (int j = 0; j < 4; ++j)
#pragma unroll
      for (int r = 0; r < 8; ++r)
        h[(size_t)(row0 + rb + r) * 512 + (ct0 + j) * 16 + col] = acc[j][r];
  }
}

// ---------------------------------------------------------------------------
// GroupNorm: stats (one block per (b,group)), then apply(+leaky[+neighbor max])
// cgShift = log2(C/G) (always a power of two here)
// ---------------------------------------------------------------------------
__global__ void gn_stats(const float* __restrict__ h, float* __restrict__ stats,
                         int S, int C, int G, int cgShift) {
  const int bg = blockIdx.x;
  const int b = bg / G, g = bg % G;
  const int Cg = 1 << cgShift;
  const long total = (long)S << cgShift;
  float sum = 0.f, sq = 0.f;
  for (long e = threadIdx.x; e < total; e += blockDim.x) {
    int s = (int)(e >> cgShift);
    int c = (g << cgShift) + (int)(e & (Cg - 1));
    float v = h[((size_t)b * S + s) * C + c];
    sum += v; sq += v * v;
  }
  __shared__ float ssum[256], ssq[256];
  ssum[threadIdx.x] = sum; ssq[threadIdx.x] = sq;
  __syncthreads();
  for (int off = 128; off > 0; off >>= 1) {
    if ((int)threadIdx.x < off) {
      ssum[threadIdx.x] += ssum[threadIdx.x + off];
      ssq[threadIdx.x]  += ssq[threadIdx.x + off];
    }
    __syncthreads();
  }
  if (threadIdx.x == 0) {
    float mean = ssum[0] / (float)total;
    float var  = ssq[0] / (float)total - mean * mean;
    stats[bg * 2]     = mean;
    stats[bg * 2 + 1] = rsqrtf(var + 1e-5f);
  }
}

__global__ void gn_apply_max(const float* __restrict__ h, const float* __restrict__ stats,
                             const float* __restrict__ gamma, const float* __restrict__ beta,
                             float* __restrict__ xout, int C, int G) {
  int t = blockIdx.x * blockDim.x + threadIdx.x;
  if (t >= BB * NPT * C) return;
  int c = t % C;
  size_t bn = (size_t)(t / C);
  int b = (int)(bn / NPT);
  int g = c / (C / G);
  float mean = stats[(b * G + g) * 2];
  float inv  = stats[(b * G + g) * 2 + 1];
  float ga = gamma[c], be = beta[c];
  float mx = -3.4e38f;
  for (int kk = 0; kk < KK; ++kk) {
    float v = h[(bn * KK + kk) * (size_t)C + c];
    v = (v - mean) * inv * ga + be;
    v = v > 0.f ? v : 0.2f * v;
    mx = fmaxf(mx, v);
  }
  xout[bn * (size_t)C + c] = mx;
}

__global__ void gn_apply_leaky(float* __restrict__ h, const float* __restrict__ stats,
                               const float* __restrict__ gamma, const float* __restrict__ beta,
                               int C, int G) {
  int t = blockIdx.x * blockDim.x + threadIdx.x;
  if (t >= BB * NPT * C) return;
  int c = t % C;
  int bn = t / C;
  int b = bn / NPT;
  int g = c / (C / G);
  float mean = stats[(b * G + g) * 2];
  float inv  = stats[(b * G + g) * 2 + 1];
  float v = h[t];
  v = (v - mean) * inv * gamma[c] + beta[c];
  h[t] = v > 0.f ? v : 0.2f * v;
}

// feat[b, 0:512] = max_n, feat[b, 512:1024] = mean_n
__global__ void pool_feat(const float* __restrict__ hn, float* __restrict__ feat) {
  const int bc = blockIdx.x;
  const int b = bc / 512, c = bc % 512;
  float mx = -3.4e38f, sm = 0.f;
  for (int n = threadIdx.x; n < NPT; n += blockDim.x) {
    float v = hn[((size_t)b * NPT + n) * 512 + c];
    mx = fmaxf(mx, v); sm += v;
  }
  __shared__ float smx[256], ssm[256];
  smx[threadIdx.x] = mx; ssm[threadIdx.x] = sm;
  __syncthreads();
  for (int off = 128; off > 0; off >>= 1) {
    if ((int)threadIdx.x < off) {
      smx[threadIdx.x] = fmaxf(smx[threadIdx.x], smx[threadIdx.x + off]);
      ssm[threadIdx.x] += ssm[threadIdx.x + off];
    }
    __syncthreads();
  }
  if (threadIdx.x == 0) {
    feat[b * 1024 + c]       = smx[0];
    feat[b * 1024 + 512 + c] = ssm[0] / (float)NPT;
  }
}

__global__ void mu_lv_kernel(const float* __restrict__ feat,
                             const float* __restrict__ muW, const float* __restrict__ muB,
                             const float* __restrict__ lvW, const float* __restrict__ lvB,
                             float* __restrict__ z, float* __restrict__ omu, float* __restrict__ olv) {
  int t = blockIdx.x * blockDim.x + threadIdx.x;
  if (t >= BB * ZD) return;
  int b = t / ZD, c = t % ZD;
  float m = muB[c], l = lvB[c];
  const float* f  = feat + b * CD;
  const float* wm = muW + (size_t)c * CD;
  const float* wl = lvW + (size_t)c * CD;
  for (int i = 0; i < CD; ++i) { m += wm[i] * f[i]; l += wl[i] * f[i]; }
  z[t] = m; omu[t] = m; olv[t] = l;
}

// Per-batch constant part of fold/head layer-0: cb[b,c] = bias[c] + W[c,3:]·[z|feat]
__global__ void fold_cb(const float* __restrict__ W, const float* __restrict__ bias,
                        const float* __restrict__ z, const float* __restrict__ feat,
                        float* __restrict__ cb, int OutC) {
  int t = blockIdx.x * blockDim.x + threadIdx.x;
  if (t >= BB * OutC) return;
  int b = t / OutC, c = t % OutC;
  const float* w = W + (size_t)c * 1539 + 3;
  float acc = bias[c];
  const float* zb = z + b * ZD;
  for (int i = 0; i < ZD; ++i) acc += w[i] * zb[i];
  const float* fb = feat + b * CD;
  for (int i = 0; i < CD; ++i) acc += w[ZD + i] * fb[i];
  cb[t] = acc;
}

// ---------------------------------------------------------------------------
// Fold MLP: 3 -> 512 -> 512 -> 256 -> 3 (relu between), WMMA for 512x512/512x256
// One wave per 16-point tile; bf16 activations ping-pong in LDS (padded pitch).
// ---------------------------------------------------------------------------
__global__ void fold_mlp(const float* __restrict__ inpos, int in_is_tmpl,
                         const float* __restrict__ W0, const float* __restrict__ cb0,
                         const unsigned int* __restrict__ pkW1, const float* __restrict__ b1,
                         const unsigned int* __restrict__ pkW2, const float* __restrict__ b2,
                         const float* __restrict__ W3, const float* __restrict__ b3,
                         float* __restrict__ outpos) {
  constexpr int P512 = 520;   // 512 + 8 pad
  constexpr int P256 = 264;   // 256 + 8 pad
  __shared__ unsigned short sA[16 * P512];
  __shared__ unsigned short sB[16 * P512];
  __shared__ float t3[16 * 3];
  const int lane = threadIdx.x;
  const int row0 = blockIdx.x * 16;        // in [0, BB*NF)
  const int b = row0 / NF;

  for (int e = lane; e < 48; e += 32) {
    int m = e / 3, d = e % 3;
    int r = row0 + m;
    int ri = in_is_tmpl ? (r % NF) : r;
    t3[m * 3 + d] = inpos[(size_t)ri * 3 + d];
  }
  __syncthreads();

  // layer0: inner dim 3 + per-batch constant bias, relu
  for (int e = lane; e < 16 * 512; e += 32) {
    int m = e >> 9, c = e & 511;
    const float* w = W0 + (size_t)c * 1539;
    float v = cb0[b * 512 + c] + t3[m*3+0]*w[0] + t3[m*3+1]*w[1] + t3[m*3+2]*w[2];
    sA[m * P512 + c] = f2bf(v > 0.f ? v : 0.f);
  }
  __syncthreads();

  // layer1: 512 -> 512 (WMMA) + relu -> sB
  for (int ct0 = 0; ct0 < 32; ct0 += 4) {
    if (ct0 + 4 < 32)
      __builtin_prefetch((const void*)(pkW1 + ((size_t)(ct0 + 4) * 16) * 256), 0, 1);
    v8f acc[4] = {};
    gemm_group4(sA, P512, pkW1, ct0, 16, lane, acc);
    int col = lane & 15, rb = (lane < 16) ? 0 : 8;
#pragma unroll
    for (int j = 0; j < 4; ++j) {
      float bv = b1[(ct0 + j) * 16 + col];
#pragma unroll
      for (int r = 0; r < 8; ++r) {
        float v = acc[j][r] + bv;
        sB[(rb + r) * P512 + (ct0 + j) * 16 + col] = f2bf(v > 0.f ? v : 0.f);
      }
    }
  }
  __syncthreads();

  // layer2: 512 -> 256 (WMMA) + relu -> sA (pitch P256)
  for (int ct0 = 0; ct0 < 16; ct0 += 4) {
    if (ct0 + 4 < 16)
      __builtin_prefetch((const void*)(pkW2 + ((size_t)(ct0 + 4) * 16) * 256), 0, 1);
    v8f acc[4] = {};
    gemm_group4(sB, P512, pkW2, ct0, 16, lane, acc);
    int col = lane & 15, rb = (lane < 16) ? 0 : 8;
#pragma unroll
    for (int j = 0; j < 4; ++j) {
      float bv = b2[(ct0 + j) * 16 + col];
#pragma unroll
      for (int r = 0; r < 8; ++r) {
        float v = acc[j][r] + bv;
        sA[(rb + r) * P256 + (ct0 + j) * 16 + col] = f2bf(v > 0.f ? v : 0.f);
      }
    }
  }
  __syncthreads();

  // layer3: 256 -> 3
#pragma unroll
  for (int pass = 0; pass < 2; ++pass) {
    int comp = (lane >> 4) + 2 * pass;
    if (comp < 3) {
      int m = lane & 15;
      float acc = b3[comp];
      for (int d = 0; d < 256; ++d)
        acc += bf2f(sA[m * P256 + d]) * W3[(size_t)comp * 256 + d];
      outpos[(size_t)(row0 + m) * 3 + comp] = acc;
    }
  }
}

// Head MLP: 3 -> 256 -> 128 -> 1 (relu, relu, sigmoid), WMMA for 256x128
__global__ void head_mlp(const float* __restrict__ pos,
                         const float* __restrict__ W0, const float* __restrict__ cb0,
                         const unsigned int* __restrict__ pkW1, const float* __restrict__ b1,
                         const float* __restrict__ W2, const float* __restrict__ b2,
                         float* __restrict__ out) {
  constexpr int P256 = 264;
  constexpr int P128 = 136;
  __shared__ unsigned short sA[16 * P256];
  __shared__ unsigned short sB[16 * P128];
  __shared__ float t3[48];
  const int lane = threadIdx.x;
  const int row0 = blockIdx.x * 16;
  const int b = row0 / NF;

  for (int e = lane; e < 48; e += 32) {
    int m = e / 3, d = e % 3;
    t3[m * 3 + d] = pos[(size_t)(row0 + m) * 3 + d];
  }
  __syncthreads();

  for (int e = lane; e < 16 * 256; e += 32) {
    int m = e >> 8, c = e & 255;
    const float* w = W0 + (size_t)c * 1539;
    float v = cb0[b * 256 + c] + t3[m*3+0]*w[0] + t3[m*3+1]*w[1] + t3[m*3+2]*w[2];
    sA[m * P256 + c] = f2bf(v > 0.f ? v : 0.f);
  }
  __syncthreads();

  for (int ct0 = 0; ct0 < 8; ct0 += 4) {
    v8f acc[4] = {};
    gemm_group4(sA, P256, pkW1, ct0, 8, lane, acc);
    int col = lane & 15, rb = (lane < 16) ? 0 : 8;
#pragma unroll
    for (int j = 0; j < 4; ++j) {
      float bv = b1[(ct0 + j) * 16 + col];
#pragma unroll
      for (int r = 0; r < 8; ++r) {
        float v = acc[j][r] + bv;
        sB[(rb + r) * P128 + (ct0 + j) * 16 + col] = f2bf(v > 0.f ? v : 0.f);
      }
    }
  }
  __syncthreads();

  if (lane < 16) {
    float acc = b2[0];
    for (int d = 0; d < 128; ++d)
      acc += bf2f(sB[lane * P128 + d]) * W2[d];
    out[row0 + lane] = 1.f / (1.f + __expf(-acc));
  }
}

// ---------------------------------------------------------------------------
// Host orchestration
// Input order (depth-first, dict insertion order):
//  0 surf, 1 template,
//  2-4 ec1 w/g/b, 5-7 ec2, 8-10 ec3, 11-13 ec4, 14-16 agg w/g/b,
//  17-18 mu W/b, 19-20 lv W/b,
//  21-28 fold1 (W,b)x4, 29-36 fold2 (W,b)x4,
//  37-42 sizing (W,b)x3, 43-48 material (W,b)x3
// Output: pos2 | sz | mat | mu | logvar (flat, fp32)
// ---------------------------------------------------------------------------
extern "C" void kernel_launch(void* const* d_in, const int* in_sizes, int n_in,
                              void* d_out, int out_size, void* d_ws, size_t ws_size,
                              hipStream_t stream) {
  auto F = [&](int i) { return (const float*)d_in[i]; };
  const float* surf = F(0);
  const float* tmpl = F(1);

  char* wsb = (char*)d_ws;
  size_t off = 0;
  auto alloc = [&](size_t bytes) -> void* {
    void* p = (void*)(wsb + off);
    off += (bytes + 255) & ~(size_t)255;
    return p;
  };
  auto pkAlloc = [&](int OutC, int KT) {
    return (unsigned int*)alloc((size_t)(OutC / 16) * KT * 256 * 4);
  };

  unsigned int* pk_ec1  = pkAlloc(64, 1);
  unsigned int* pk_ec2  = pkAlloc(128, 4);
  unsigned int* pk_ec3  = pkAlloc(256, 8);
  unsigned int* pk_ec4  = pkAlloc(512, 16);
  unsigned int* pk_agg  = pkAlloc(512, 30);
  unsigned int* pk_f1w1 = pkAlloc(512, 16);
  unsigned int* pk_f1w2 = pkAlloc(256, 16);
  unsigned int* pk_f2w1 = pkAlloc(512, 16);
  unsigned int* pk_f2w2 = pkAlloc(256, 16);
  unsigned int* pk_szw1 = pkAlloc(128, 8);
  unsigned int* pk_mtw1 = pkAlloc(128, 8);

  int*   idx   = (int*)alloc((size_t)BB * NPT * KK * 4);
  float* x1    = (float*)alloc((size_t)BB * NPT * 64 * 4);
  float* x2    = (float*)alloc((size_t)BB * NPT * 128 * 4);
  float* x3    = (float*)alloc((size_t)BB * NPT * 256 * 4);
  float* x4    = (float*)alloc((size_t)BB * NPT * 512 * 4);
  float* hbuf  = (float*)alloc((size_t)BB * NPT * KK * 512 * 4);
  float* hagg  = (float*)alloc((size_t)BB * NPT * 512 * 4);
  float* stats = (float*)alloc(BB * 32 * 2 * 4);
  float* feat  = (float*)alloc(BB * CD * 4);
  float* zbuf  = (float*)alloc(BB * ZD * 4);
  float* cb_f1 = (float*)alloc(BB * 512 * 4);
  float* cb_f2 = (float*)alloc(BB * 512 * 4);
  float* cb_sz = (float*)alloc(BB * 256 * 4);
  float* cb_mt = (float*)alloc(BB * 256 * 4);
  float* pos1  = (float*)alloc((size_t)BB * NF * 3 * 4);

  float* out      = (float*)d_out;
  float* out_pos2 = out;
  float* out_sz   = out_pos2 + (size_t)BB * NF * 3;
  float* out_mat  = out_sz + (size_t)BB * NF;
  float* out_mu   = out_mat + (size_t)BB * NF;
  float* out_lv   = out_mu + (size_t)BB * ZD;

  auto pack = [&](const float* W, unsigned int* pk, int OutC, int InC, int KT) {
    int total = (OutC / 16) * KT * 256;
    pack_weights<<<(total + 255) / 256, 256, 0, stream>>>(W, pk, OutC, InC, KT);
  };
  pack(F(2),  pk_ec1, 64, 12, 1);
  pack(F(5),  pk_ec2, 128, 128, 4);
  pack(F(8),  pk_ec3, 256, 256, 8);
  pack(F(11), pk_ec4, 512, 512, 16);
  pack(F(14), pk_agg, 512, 960, 30);
  pack(F(23), pk_f1w1, 512, 512, 16);
  pack(F(25), pk_f1w2, 256, 512, 16);
  pack(F(31), pk_f2w1, 512, 512, 16);
  pack(F(33), pk_f2w2, 256, 512, 16);
  pack(F(39), pk_szw1, 128, 256, 8);
  pack(F(45), pk_mtw1, 128, 256, 8);

  const int rowTiles = BB * NPT * KK / 16;   // 10240
  const int knnBlks  = BB * NPT / 32;        // 256

  // EdgeConv 1 (D=6 -> 64, G=8, Cg=8)
  knn_kernel<6><<<knnBlks, 32, 0, stream>>>(surf, idx);
  edgeconv_gemm<6, 64><<<rowTiles, 32, 0, stream>>>(surf, idx, pk_ec1, hbuf);
  gn_stats<<<BB * 8, 256, 0, stream>>>(hbuf, stats, NPT * KK, 64, 8, 3);
  gn_apply_max<<<(BB * NPT * 64 + 255) / 256, 256, 0, stream>>>(hbuf, stats, F(3), F(4), x1, 64, 8);
  // EdgeConv 2 (64 -> 128, G=8, Cg=16)
  knn_kernel<64><<<knnBlks, 32, 0, stream>>>(x1, idx);
  edgeconv_gemm<64, 128><<<rowTiles, 32, 0, stream>>>(x1, idx, pk_ec2, hbuf);
  gn_stats<<<BB * 8, 256, 0, stream>>>(hbuf, stats, NPT * KK, 128, 8, 4);
  gn_apply_max<<<(BB * NPT * 128 + 255) / 256, 256, 0, stream>>>(hbuf, stats, F(6), F(7), x2, 128, 8);
  // EdgeConv 3 (128 -> 256, G=16, Cg=16)
  knn_kernel<128><<<knnBlks, 32, 0, stream>>>(x2, idx);
  edgeconv_gemm<128, 256><<<rowTiles, 32, 0, stream>>>(x2, idx, pk_ec3, hbuf);
  gn_stats<<<BB * 16, 256, 0, stream>>>(hbuf, stats, NPT * KK, 256, 16, 4);
  gn_apply_max<<<(BB * NPT * 256 + 255) / 256, 256, 0, stream>>>(hbuf, stats, F(9), F(10), x3, 256, 16);
  // EdgeConv 4 (256 -> 512, G=32, Cg=16)
  knn_kernel<256><<<knnBlks, 32, 0, stream>>>(x3, idx);
  edgeconv_gemm<256, 512><<<rowTiles, 32, 0, stream>>>(x3, idx, pk_ec4, hbuf);
  gn_stats<<<BB * 32, 256, 0, stream>>>(hbuf, stats, NPT * KK, 512, 32, 4);
  gn_apply_max<<<(BB * NPT * 512 + 255) / 256, 256, 0, stream>>>(hbuf, stats, F(12), F(13), x4, 512, 32);

  // Aggregation + GN + leaky + pooling + mu/logvar
  agg_gemm<<<BB * NPT / 16, 32, 0, stream>>>(x1, x2, x3, x4, pk_agg, hagg);
  gn_stats<<<BB * 32, 256, 0, stream>>>(hagg, stats, NPT, 512, 32, 4);
  gn_apply_leaky<<<(BB * NPT * 512 + 255) / 256, 256, 0, stream>>>(hagg, stats, F(15), F(16), 512, 32);
  pool_feat<<<BB * 512, 256, 0, stream>>>(hagg, feat);
  mu_lv_kernel<<<(BB * ZD + 255) / 256, 256, 0, stream>>>(feat, F(17), F(18), F(19), F(20),
                                                          zbuf, out_mu, out_lv);

  // Per-batch layer-0 constants for decoders
  fold_cb<<<(BB * 512 + 255) / 256, 256, 0, stream>>>(F(21), F(22), zbuf, feat, cb_f1, 512);
  fold_cb<<<(BB * 512 + 255) / 256, 256, 0, stream>>>(F(29), F(30), zbuf, feat, cb_f2, 512);
  fold_cb<<<(BB * 256 + 255) / 256, 256, 0, stream>>>(F(37), F(38), zbuf, feat, cb_sz, 256);
  fold_cb<<<(BB * 256 + 255) / 256, 256, 0, stream>>>(F(43), F(44), zbuf, feat, cb_mt, 256);

  // Folding decoders + heads
  const int foldBlks = BB * NF / 16;  // 1024
  fold_mlp<<<foldBlks, 32, 0, stream>>>(tmpl, 1, F(21), cb_f1, pk_f1w1, F(24), pk_f1w2, F(26),
                                        F(27), F(28), pos1);
  fold_mlp<<<foldBlks, 32, 0, stream>>>(pos1, 0, F(29), cb_f2, pk_f2w1, F(32), pk_f2w2, F(34),
                                        F(35), F(36), out_pos2);
  head_mlp<<<foldBlks, 32, 0, stream>>>(out_pos2, F(37), cb_sz, pk_szw1, F(40), F(41), F(42), out_sz);
  head_mlp<<<foldBlks, 32, 0, stream>>>(out_pos2, F(43), cb_mt, pk_mtw1, F(46), F(47), F(48), out_mat);

  (void)in_sizes; (void)n_in; (void)out_size; (void)ws_size;
}
